// IRCA_40381282517380
// MI455X (gfx1250) — matrix operand order repaired and checked by reading, not compiled, not run on hardware
//
#include <hip/hip_runtime.h>
#include <hip/hip_bf16.h>
#include <math.h>

// ---------------------------------------------------------------------------
// MI455X (gfx1250) implementation.
// Dominant cost: 131072x1024 dot-product matrix (K=256) for k-means
// assignment = 68.7 GFLOP. Done with v_wmma_f32_16x16x32_bf16 (wave32).
// HBM floor: 128 MB of x @ 23.3 TB/s ~ 5.5us; bf16 WMMA keeps compute in the
// same ballpark instead of 100s of us in scalar f32 VALU.
// ---------------------------------------------------------------------------

typedef __attribute__((ext_vector_type(16))) __bf16 v16bf;
typedef __attribute__((ext_vector_type(8)))  float  v8f;

union BF16Frag { v16bf v; unsigned int u[8]; };

__device__ __forceinline__ float waveReduceSum(float x) {
#pragma unroll
  for (int off = 16; off > 0; off >>= 1) x += __shfl_xor(x, off, 32);
  return x;
}

// ---- zero the accumulators (harness poisons ws with 0xAA) ------------------
__global__ void zero_ws_kernel(float* __restrict__ sums,
                               unsigned int* __restrict__ counts) {
  int i = blockIdx.x * blockDim.x + threadIdx.x;
  int stride = gridDim.x * blockDim.x;
  for (int j = i; j < 1024 * 256; j += stride) sums[j] = 0.0f;
  for (int j = i; j < 1024; j += stride) counts[j] = 0u;
}

// ---- l2-normalize means rows -> f32 (fallback copy) + bf16 (WMMA B) --------
__global__ void prep_means_kernel(const float* __restrict__ means,
                                  float* __restrict__ meansN,
                                  __bf16* __restrict__ meansBf) {
  const int lane = threadIdx.x & 31;
  const int w    = threadIdx.x >> 5;
  const int row  = blockIdx.x * 8 + w;           // 128 blocks * 8 waves = 1024
  const float* src = means + (size_t)row * 256;
  float vals[8]; float ss = 0.0f;
#pragma unroll
  for (int i = 0; i < 8; ++i) { vals[i] = src[lane + 32 * i]; ss += vals[i] * vals[i]; }
  ss = waveReduceSum(ss);
  const float inv = 1.0f / fmaxf(sqrtf(ss), 1e-12f);
#pragma unroll
  for (int i = 0; i < 8; ++i) {
    const float y = vals[i] * inv;
    const size_t idx = (size_t)row * 256 + lane + 32 * i;
    meansN[idx]  = y;
    meansBf[idx] = (__bf16)y;
  }
}

// ---- convert projection weights to bf16 ------------------------------------
__global__ void convert_w_kernel(const float* __restrict__ wk,
                                 const float* __restrict__ wv,
                                 __bf16* __restrict__ wkBf,
                                 __bf16* __restrict__ wvBf) {
  int i = blockIdx.x * blockDim.x + threadIdx.x;
  int stride = gridDim.x * blockDim.x;
  for (int j = i; j < 65536; j += stride) {
    wkBf[j] = (__bf16)wk[j];
    wvBf[j] = (__bf16)wv[j];
  }
}

// ---- main kernel: normalize rows, WMMA argmax vs 1024 means, scatter-add ---
// Block = 256 threads = 8 waves; each wave owns a 16-row tile (128 rows/block,
// grid 1024 => 131072 rows). Means tile (16 cols x 256 K, bf16 = 8KB) is staged
// in LDS once per block and shared by all 8 waves.
__global__ __launch_bounds__(256) void assign_scatter_kernel(
    const float* __restrict__ x,              // [131072, 256]
    const unsigned int* __restrict__ meansBfW,// [1024,256] bf16 viewed as dwords
    float* __restrict__ sums,                 // [1024, 256]
    unsigned int* __restrict__ counts) {      // [1024]
  __shared__ __bf16       ldsX[8][16 * 256];  // 64 KB: per-wave A tiles
  __shared__ unsigned int ldsB[16 * 128];     // 8 KB: shared means tile (dwords)
  __shared__ float        ldsInv[128];
  __shared__ int          ldsBucket[128];

  const int tid  = threadIdx.x;
  const int lane = tid & 31;
  const int w    = tid >> 5;
  const int hi   = lane >> 4;   // which 16-lane half
  const int lo   = lane & 15;
  const int rowBase = blockIdx.x * 128 + w * 16;

  // Phase 1: load + l2-normalize 16 rows into the wave's LDS bf16 tile.
  for (int rr = 0; rr < 16; ++rr) {
    const float* xr = x + (size_t)(rowBase + rr) * 256;
    float vals[8]; float ss = 0.0f;
#pragma unroll
    for (int i = 0; i < 8; ++i) { vals[i] = xr[lane + 32 * i]; ss += vals[i] * vals[i]; }
    ss = waveReduceSum(ss);
    const float inv = 1.0f / fmaxf(sqrtf(ss), 1e-12f);
#pragma unroll
    for (int i = 0; i < 8; ++i)
      ldsX[w][rr * 256 + lane + 32 * i] = (__bf16)(vals[i] * inv);
    if (lane == 0) ldsInv[w * 16 + rr] = inv;
  }
  __syncthreads();

  // Build A fragments (16x32 bf16 per K-chunk), held in registers for all
  // 64 column tiles. ISA A layout: lane row = lo; lanes0-15 K 0-7/16-23,
  // lanes16-31 K 8-15/24-31 -> dwords base + {0..3} and base+8 + {0..3}.
  BF16Frag a[8];
  {
    const unsigned int* xw = (const unsigned int*)&ldsX[w][0];
#pragma unroll
    for (int kc = 0; kc < 8; ++kc) {
      const int base = lo * 128 + kc * 16 + hi * 4;
#pragma unroll
      for (int j = 0; j < 4; ++j) {
        a[kc].u[j]     = xw[base + j];
        a[kc].u[4 + j] = xw[base + 8 + j];
      }
    }
  }

  float bestVal[8]; int bestIdx[8];
#pragma unroll
  for (int r = 0; r < 8; ++r) { bestVal[r] = -3.4e38f; bestIdx[r] = 0; }

  // Phase 2: sweep 64 column tiles of 16 means; 8 K-chunks each.
  for (int tb = 0; tb < 64; ++tb) {
    __syncthreads();
    const unsigned int* mb = meansBfW + (size_t)tb * 2048; // 16 contiguous rows
    for (int i = tid; i < 2048; i += 256) ldsB[i] = mb[i];
    __syncthreads();

    v8f c = {};
#pragma unroll
    for (int kc = 0; kc < 8; ++kc) {
      // B layout (same K order as A): lane column = lo; lanes0-15 K 0-7/16-23,
      // lanes16-31 K 8-15/24-31 -> dwords base + {0..3} and base+8 + {0..3}.
      BF16Frag b;
      const int base = lo * 128 + kc * 16 + hi * 4;
#pragma unroll
      for (int j = 0; j < 4; ++j) {
        b.u[j]     = ldsB[base + j];
        b.u[4 + j] = ldsB[base + 8 + j];
      }
      c = __builtin_amdgcn_wmma_f32_16x16x32_bf16(
              false, a[kc].v, false, b.v, (short)0, c, false, false);
    }

    // C layout: VGPR r holds row (r + 8*hi), column = lo within tile.
    const int t = tb * 16 + lo;
#pragma unroll
    for (int r = 0; r < 8; ++r)
      if (c[r] > bestVal[r]) { bestVal[r] = c[r]; bestIdx[r] = t; }
  }

  // Cross-lane argmax inside each 16-lane half (prefer smaller index on tie,
  // matching jnp.argmax first-occurrence semantics).
#pragma unroll
  for (int r = 0; r < 8; ++r) {
#pragma unroll
    for (int off = 8; off > 0; off >>= 1) {
      const float ov = __shfl_xor(bestVal[r], off, 32);
      const int   oi = __shfl_xor(bestIdx[r], off, 32);
      if (ov > bestVal[r] || (ov == bestVal[r] && oi < bestIdx[r])) {
        bestVal[r] = ov; bestIdx[r] = oi;
      }
    }
  }
  if (lo == 0) {
#pragma unroll
    for (int r = 0; r < 8; ++r) ldsBucket[w * 16 + r + 8 * hi] = bestIdx[r];
  }
  __syncthreads();

  // Phase 3: scatter-add normalized f32 rows into bucket sums (L2 atomics).
  for (int rr = 0; rr < 16; ++rr) {
    const int   b   = ldsBucket[w * 16 + rr];
    const float inv = ldsInv[w * 16 + rr];
    const float* xr = x + (size_t)(rowBase + rr) * 256;
#pragma unroll
    for (int i = 0; i < 8; ++i) {
      const int d = lane + 32 * i;
      atomicAdd(&sums[(size_t)b * 256 + d], xr[d] * inv);
    }
    if (lane == 0) atomicAdd(&counts[b], 1u);
  }
}

// ---- finalize: x_global = counts ? l2norm(sums) : old normalized mean ------
__global__ void finalize_means_kernel(const float* __restrict__ sums,
                                      const unsigned int* __restrict__ counts,
                                      const float* __restrict__ meansN,
                                      float* __restrict__ xg,      // d_out[2]
                                      __bf16* __restrict__ xgBf) {
  const int lane = threadIdx.x & 31;
  const int w    = threadIdx.x >> 5;
  const int row  = blockIdx.x * 8 + w;          // 128 blocks -> 1024 rows
  const unsigned int cnt = counts[row];
  const float* src = cnt ? (sums + (size_t)row * 256) : (meansN + (size_t)row * 256);
  float vals[8]; float ss = 0.0f;
#pragma unroll
  for (int i = 0; i < 8; ++i) { vals[i] = src[lane + 32 * i]; ss += vals[i] * vals[i]; }
  ss = waveReduceSum(ss);
  const float inv = cnt ? (1.0f / fmaxf(sqrtf(ss), 1e-12f)) : 1.0f;
#pragma unroll
  for (int i = 0; i < 8; ++i) {
    const float y = vals[i] * inv;
    const size_t idx = (size_t)row * 256 + lane + 32 * i;
    xg[idx]   = y;
    xgBf[idx] = (__bf16)y;
  }
}

// ---- k/v projections: x_global @ W^T via bf16 WMMA, write [H,T,dh] layout --
// 2048 tiles = 2 matrices * 64 t-tiles * 16 o-tiles; one wave per tile.
__global__ __launch_bounds__(256) void proj_kv_kernel(
    const unsigned int* __restrict__ xgW,  // [1024,256] bf16 as dwords
    const unsigned int* __restrict__ wkW,  // [256,256] bf16 as dwords
    const unsigned int* __restrict__ wvW,
    float* __restrict__ outK, float* __restrict__ outV) {
  const int lane = threadIdx.x & 31;
  const int w    = threadIdx.x >> 5;
  const int hi   = lane >> 4;
  const int lo   = lane & 15;
  const int tile = blockIdx.x * 8 + w;      // 256 blocks -> 2048 tiles
  const int mat  = tile >> 10;
  const int rem  = tile & 1023;
  const int t0   = (rem >> 4) * 16;
  const int o0   = (rem & 15) * 16;
  const unsigned int* W = mat ? wvW : wkW;
  float* outp = mat ? outV : outK;

  v8f c = {};
#pragma unroll
  for (int kc = 0; kc < 8; ++kc) {
    BF16Frag A, B;
    const int abase = (t0 + lo) * 128 + kc * 16 + hi * 4;
#pragma unroll
    for (int j = 0; j < 4; ++j) {
      A.u[j]     = xgW[abase + j];
      A.u[4 + j] = xgW[abase + 8 + j];
    }
    const int bbase = (o0 + lo) * 128 + kc * 16 + hi * 4;
#pragma unroll
    for (int j = 0; j < 4; ++j) {
      B.u[j]     = W[bbase + j];
      B.u[4 + j] = W[bbase + 8 + j];
    }
    c = __builtin_amdgcn_wmma_f32_16x16x32_bf16(
            false, A.v, false, B.v, (short)0, c, false, false);
  }

  // Output k[h][t][j]: o = o0+lo, flat = (o/32)*32768 + t*32 + (o%32).
  const int o = o0 + lo;
  const int obase = (o >> 5) * 32768 + (o & 31);
#pragma unroll
  for (int r = 0; r < 8; ++r) {
    const int t = t0 + r + 8 * hi;
    outp[obase + t * 32] = c[r];
  }
}

// ---------------------------------------------------------------------------
extern "C" void kernel_launch(void* const* d_in, const int* in_sizes, int n_in,
                              void* d_out, int out_size, void* d_ws, size_t ws_size,
                              hipStream_t stream) {
  const float* x     = (const float*)d_in[0];  // [8,16384,256]
  const float* means = (const float*)d_in[1];  // [1024,256]
  const float* wk    = (const float*)d_in[2];  // [256,256]
  const float* wv    = (const float*)d_in[3];  // [256,256]

  char* ws = (char*)d_ws;                      // ~3.3 MB used
  float*        meansN  = (float*)(ws + 0);          // 1 MB
  float*        sums    = (float*)(ws + 1048576);    // 1 MB
  __bf16*       meansBf = (__bf16*)(ws + 2097152);   // 512 KB
  __bf16*       xgBf    = (__bf16*)(ws + 2621440);   // 512 KB
  __bf16*       wkBf    = (__bf16*)(ws + 3145728);   // 128 KB
  __bf16*       wvBf    = (__bf16*)(ws + 3276800);   // 128 KB
  unsigned int* counts  = (unsigned int*)(ws + 3407872); // 4 KB

  float* outK  = (float*)d_out;        // [8,1024,32] = 262144
  float* outV  = outK + 262144;        // [8,1024,32]
  float* outXg = outK + 524288;        // [1024,256]

  zero_ws_kernel      <<<256,  256, 0, stream>>>(sums, counts);
  prep_means_kernel   <<<128,  256, 0, stream>>>(means, meansN, meansBf);
  convert_w_kernel    <<<128,  256, 0, stream>>>(wk, wv, wkBf, wvBf);
  assign_scatter_kernel<<<1024, 256, 0, stream>>>(
      x, (const unsigned int*)meansBf, sums, counts);
  finalize_means_kernel<<<128, 256, 0, stream>>>(sums, counts, meansN, outXg, xgBf);
  proj_kv_kernel      <<<256,  256, 0, stream>>>(
      (const unsigned int*)xgBf, (const unsigned int*)wkBf,
      (const unsigned int*)wvBf, outK, outV);
}
